// MPNNModel_33346126086659
// MI455X (gfx1250) — matrix operand order repaired and checked by reading, not compiled, not run on hardware
//
#include <hip/hip_runtime.h>

// ---------------------------------------------------------------- constants
#define IN_DIM   32
#define EDGE_DIM 16
#define HID      64
#define MLP_H    32
#define N_LAYERS 5
#define WPB      8        // waves per block (256 threads, wave32)

typedef __attribute__((ext_vector_type(16))) __bf16          v16bf;
typedef __attribute__((ext_vector_type(16))) unsigned short  v16u;
typedef __attribute__((ext_vector_type(8)))  float           v8f;

// ---------------------------------------------------------------- helpers
__device__ __forceinline__ unsigned short f2bf(float f) {
    unsigned int u = __float_as_uint(f);
    unsigned int r = u + 0x7FFFu + ((u >> 16) & 1u);   // round-to-nearest-even
    return (unsigned short)(r >> 16);
}

__device__ __forceinline__ void fill8(v16u& v, int off, const unsigned short* p) {
    union { uint4 q; unsigned short s[8]; } u;
    u.q = *(const uint4*)p;                              // 16B aligned load
#pragma unroll
    for (int j = 0; j < 8; ++j) v[off + j] = u.s[j];
}

__device__ __forceinline__ v16bf bcast_bf(v16u v) {
    return __builtin_bit_cast(v16bf, v);
}

// packed B fragment: 16 contiguous ushorts per lane
__device__ __forceinline__ v16bf load_b_frag(const unsigned short* p) {
    v16u v; fill8(v, 0, p); fill8(v, 8, p + 8);
    return __builtin_bit_cast(v16bf, v);
}

__device__ __forceinline__ v8f wmma_bf16(v16bf a, v16bf b, v8f c) {
    // (neg_a, A, neg_b, B, c_mod, C, reuse_a, reuse_b)
    return __builtin_amdgcn_wmma_f32_16x16x32_bf16(false, a, false, b, (short)0, c, false, false);
}

// ---------------------------------------------------------------- weight packing
// Output layout: frag-major [(chunk*NT + nt)][lane][j], NT = N/16,
// lane n = nt*16 + (lane&15), k = chunk*32 + (j<8 ? (lane>>4)*8 + j : 16 + (lane>>4)*8 + j-8)  (zero-pad k>=Ka)
__global__ void pack_w_kernel(const float* __restrict__ W, unsigned short* __restrict__ out,
                              int Ka, int N, int total) {
    int idx = blockIdx.x * blockDim.x + threadIdx.x;
    if (idx >= total) return;
    int NT    = N >> 4;
    int frag  = idx >> 9;
    int lane  = (idx >> 4) & 31;
    int j     = idx & 15;
    int chunk = frag / NT;
    int nt    = frag - chunk * NT;
    int k     = chunk * 32 + ((lane >> 4) << 3) + (j < 8 ? j : 8 + j);
    int n     = nt * 16 + (lane & 15);
    out[idx]  = (k < Ka) ? f2bf(W[(size_t)k * N + n]) : (unsigned short)0;
}

// ---------------------------------------------------------------- h0 = x @ Win + b
__global__ void lin_in_kernel(const float* __restrict__ x, const float* __restrict__ W,
                              const float* __restrict__ b, float* __restrict__ h,
                              unsigned short* __restrict__ h_bf, int n_nodes) {
    int idx = blockIdx.x * blockDim.x + threadIdx.x;
    if (idx >= n_nodes * HID) return;
    int i = idx >> 6, o = idx & 63;
    const float* xr = x + (size_t)i * IN_DIM;
    float acc = b[o];
#pragma unroll
    for (int k = 0; k < IN_DIM; ++k) acc += xr[k] * W[k * HID + o];
    h[idx]    = acc;
    h_bf[idx] = f2bf(acc);
}

// ---------------------------------------------------------------- edge message MLP (fused)
__global__ __launch_bounds__(256)
void edge_msg_kernel(const int* __restrict__ src, const int* __restrict__ dst,
                     const float* __restrict__ edge_attr,
                     const unsigned short* __restrict__ h_bf,
                     float* __restrict__ aggr,
                     const unsigned short* __restrict__ w1p,
                     const unsigned short* __restrict__ w2p,
                     const float* __restrict__ b1, const float* __restrict__ g1, const float* __restrict__ c1,
                     const float* __restrict__ b2, const float* __restrict__ g2, const float* __restrict__ c2,
                     int n_edges, int n_tiles) {
    __shared__ unsigned short smem[WPB][16][MLP_H];     // 8 KB
    const int lane = threadIdx.x & 31;
    const int wid  = threadIdx.x >> 5;
    int tile = blockIdx.x * WPB + wid;
    const bool tile_ok = tile < n_tiles;
    tile = tile_ok ? tile : (n_tiles - 1);
    const int eb   = tile * 16;
    const int mrow = lane & 15;
    const int hi   = lane >> 4;
    const int kb   = hi * 8;
    const int e    = min(eb + mrow, n_edges - 1);
    const int d    = dst[e];
    const int s    = src[e];
    const unsigned short* hD = h_bf + (size_t)d * HID;
    const unsigned short* hS = h_bf + (size_t)s * HID;

    // A fragments: K = [h[dst](64) | h[src](64) | edge_attr(16) | pad(16)]
    v16u a0u, a1u, a2u, a3u, a4u;
    fill8(a0u, 0, hD + kb);       fill8(a0u, 8, hD + 16 + kb);
    fill8(a1u, 0, hD + 32 + kb);  fill8(a1u, 8, hD + 48 + kb);
    fill8(a2u, 0, hS + kb);       fill8(a2u, 8, hS + 16 + kb);
    fill8(a3u, 0, hS + 32 + kb);  fill8(a3u, 8, hS + 48 + kb);
    const float* ea = edge_attr + (size_t)e * EDGE_DIM + kb;
#pragma unroll
    for (int j = 0; j < 8; ++j) { a4u[j] = f2bf(ea[j]); a4u[8 + j] = 0; }
    v16bf a[5] = { bcast_bf(a0u), bcast_bf(a1u), bcast_bf(a2u), bcast_bf(a3u), bcast_bf(a4u) };

    // GEMM1: [16,160] x [160,32]  -> 10 WMMAs
    v8f acc0 = {}, acc1 = {};
#pragma unroll
    for (int c = 0; c < 5; ++c) {
        v16bf bA = load_b_frag(w1p + (c * 2 + 0) * 512 + lane * 16);
        v16bf bB = load_b_frag(w1p + (c * 2 + 1) * 512 + lane * 16);
        acc0 = wmma_bf16(a[c], bA, acc0);
        acc1 = wmma_bf16(a[c], bB, acc1);
    }

    // epilogue1: relu(bn(x+b)) -> stage bf16 to LDS in row-major [16][32]
    const float rs = rsqrtf(1.0f + 1e-5f);
    {
        int n0 = lane & 15;
        float p0b = b1[n0],      p0g = g1[n0]      * rs, p0c = c1[n0];
        float p1b = b1[16 + n0], p1g = g1[16 + n0] * rs, p1c = c1[16 + n0];
#pragma unroll
        for (int r = 0; r < 8; ++r) {
            int row  = r + hi * 8;
            float v0 = (acc0[r] + p0b) * p0g + p0c; v0 = v0 > 0.f ? v0 : 0.f;
            float v1 = (acc1[r] + p1b) * p1g + p1c; v1 = v1 > 0.f ? v1 : 0.f;
            smem[wid][row][n0]      = f2bf(v0);
            smem[wid][row][16 + n0] = f2bf(v1);
        }
    }
    __syncthreads();

    // reload in A layout
    v16u mu;
#pragma unroll
    for (int j = 0; j < 8; ++j) {
        mu[j]     = smem[wid][mrow][kb + j];
        mu[8 + j] = smem[wid][mrow][16 + kb + j];
    }
    v16bf am = bcast_bf(mu);

    // GEMM2: [16,32] x [32,64] -> 4 WMMAs
    v8f acc2[4] = { v8f{}, v8f{}, v8f{}, v8f{} };
#pragma unroll
    for (int nt = 0; nt < 4; ++nt) {
        v16bf bW = load_b_frag(w2p + nt * 512 + lane * 16);
        acc2[nt] = wmma_bf16(am, bW, acc2[nt]);
    }

    // scatter-add bn(relu(x+b)) into aggr[dst]
    int dr[8];
#pragma unroll
    for (int r = 0; r < 8; ++r) dr[r] = dst[min(eb + hi * 8 + r, n_edges - 1)];
#pragma unroll
    for (int nt = 0; nt < 4; ++nt) {
        int n = nt * 16 + (lane & 15);
        float pb = b2[n], pg = g2[n] * rs, pc = c2[n];
#pragma unroll
        for (int r = 0; r < 8; ++r) {
            float v = acc2[nt][r] + pb; v = v > 0.f ? v : 0.f;
            v = v * pg + pc;
            int rowe = eb + hi * 8 + r;
            if (tile_ok && rowe < n_edges)
                atomicAdd(&aggr[(size_t)dr[r] * HID + n], v);
        }
    }
}

// ---------------------------------------------------------------- node update MLP (fused, residual)
__global__ __launch_bounds__(256)
void node_upd_kernel(float* __restrict__ h, unsigned short* __restrict__ h_bf,
                     const float* __restrict__ aggr,
                     const unsigned short* __restrict__ w1p,
                     const unsigned short* __restrict__ w2p,
                     const float* __restrict__ b1, const float* __restrict__ g1, const float* __restrict__ c1,
                     const float* __restrict__ b2, const float* __restrict__ g2, const float* __restrict__ c2,
                     int n_nodes, int n_tiles) {
    __shared__ unsigned short smem[WPB][16][MLP_H];
    const int lane = threadIdx.x & 31;
    const int wid  = threadIdx.x >> 5;
    int tile = blockIdx.x * WPB + wid;
    const bool tile_ok = tile < n_tiles;
    tile = tile_ok ? tile : (n_tiles - 1);
    const int nb   = tile * 16;
    const int mrow = lane & 15;
    const int hi   = lane >> 4;
    const int kb   = hi * 8;
    const int i    = min(nb + mrow, n_nodes - 1);
    const unsigned short* hr = h_bf + (size_t)i * HID;
    const float*          ar = aggr + (size_t)i * HID;

    // A = [h(64) | aggr(64)]
    v16u a0u, a1u, a2u, a3u;
    fill8(a0u, 0, hr + kb);       fill8(a0u, 8, hr + 16 + kb);
    fill8(a1u, 0, hr + 32 + kb);  fill8(a1u, 8, hr + 48 + kb);
#pragma unroll
    for (int j = 0; j < 8; ++j) {
        a2u[j]     = f2bf(ar[kb + j]);
        a2u[8 + j] = f2bf(ar[16 + kb + j]);
        a3u[j]     = f2bf(ar[32 + kb + j]);
        a3u[8 + j] = f2bf(ar[48 + kb + j]);
    }
    v16bf a[4] = { bcast_bf(a0u), bcast_bf(a1u), bcast_bf(a2u), bcast_bf(a3u) };

    // GEMM1: [16,128] x [128,32] -> 8 WMMAs
    v8f acc0 = {}, acc1 = {};
#pragma unroll
    for (int c = 0; c < 4; ++c) {
        v16bf bA = load_b_frag(w1p + (c * 2 + 0) * 512 + lane * 16);
        v16bf bB = load_b_frag(w1p + (c * 2 + 1) * 512 + lane * 16);
        acc0 = wmma_bf16(a[c], bA, acc0);
        acc1 = wmma_bf16(a[c], bB, acc1);
    }

    const float rs = rsqrtf(1.0f + 1e-5f);
    {
        int n0 = lane & 15;
        float p0b = b1[n0],      p0g = g1[n0]      * rs, p0c = c1[n0];
        float p1b = b1[16 + n0], p1g = g1[16 + n0] * rs, p1c = c1[16 + n0];
#pragma unroll
        for (int r = 0; r < 8; ++r) {
            int row  = r + hi * 8;
            float v0 = (acc0[r] + p0b) * p0g + p0c; v0 = v0 > 0.f ? v0 : 0.f;
            float v1 = (acc1[r] + p1b) * p1g + p1c; v1 = v1 > 0.f ? v1 : 0.f;
            smem[wid][row][n0]      = f2bf(v0);
            smem[wid][row][16 + n0] = f2bf(v1);
        }
    }
    __syncthreads();

    v16u mu;
#pragma unroll
    for (int j = 0; j < 8; ++j) {
        mu[j]     = smem[wid][mrow][kb + j];
        mu[8 + j] = smem[wid][mrow][16 + kb + j];
    }
    v16bf am = bcast_bf(mu);

    v8f acc2[4] = { v8f{}, v8f{}, v8f{}, v8f{} };
#pragma unroll
    for (int nt = 0; nt < 4; ++nt) {
        v16bf bW = load_b_frag(w2p + nt * 512 + lane * 16);
        acc2[nt] = wmma_bf16(am, bW, acc2[nt]);
    }

    // u = bn(relu(x+b)); h += u (residual); refresh bf16 mirror
#pragma unroll
    for (int nt = 0; nt < 4; ++nt) {
        int n = nt * 16 + (lane & 15);
        float pb = b2[n], pg = g2[n] * rs, pc = c2[n];
#pragma unroll
        for (int r = 0; r < 8; ++r) {
            int row = r + hi * 8;
            int i2  = nb + row;
            if (tile_ok && i2 < n_nodes) {
                float v  = acc2[nt][r] + pb; v = v > 0.f ? v : 0.f;
                v        = v * pg + pc;
                size_t o = (size_t)i2 * HID + n;
                float hv = h[o] + v;
                h[o]     = hv;
                h_bf[o]  = f2bf(hv);
            }
        }
    }
}

// ---------------------------------------------------------------- global mean pool + head
__global__ void pool_accum_kernel(const float* __restrict__ h, const int* __restrict__ batch,
                                  float* __restrict__ sums, float* __restrict__ cnt, int n_nodes) {
    int i = blockIdx.x * blockDim.x + threadIdx.x;
    if (i >= n_nodes) return;
    int g = batch[i];
    const float* hr = h + (size_t)i * HID;
#pragma unroll 4
    for (int k = 0; k < HID; ++k) atomicAdd(&sums[(size_t)g * HID + k], hr[k]);
    atomicAdd(&cnt[g], 1.0f);
}

__global__ void pool_out_kernel(const float* __restrict__ sums, const float* __restrict__ cnt,
                                const float* __restrict__ out_w, const float* __restrict__ out_b,
                                float* __restrict__ out, int n_graphs) {
    int g = blockIdx.x * blockDim.x + threadIdx.x;
    if (g >= n_graphs) return;
    float c = cnt[g]; c = c > 1.f ? c : 1.f;
    float acc = 0.f;
#pragma unroll
    for (int k = 0; k < HID; ++k) acc += (sums[(size_t)g * HID + k] / c) * out_w[k];
    out[g] = acc + out_b[0];
}

// ---------------------------------------------------------------- host driver
extern "C" void kernel_launch(void* const* d_in, const int* in_sizes, int n_in,
                              void* d_out, int out_size, void* d_ws, size_t ws_size,
                              hipStream_t stream) {
    const float* x         = (const float*)d_in[0];
    const int*   ei        = (const int*)  d_in[1];
    const float* edge_attr = (const float*)d_in[2];
    const int*   batch     = (const int*)  d_in[3];
    const float* lin_in_w  = (const float*)d_in[4];
    const float* lin_in_b  = (const float*)d_in[5];
    const float* msg_w1    = (const float*)d_in[6];
    const float* msg_b1    = (const float*)d_in[7];
    const float* msg_g1    = (const float*)d_in[8];
    const float* msg_c1    = (const float*)d_in[9];
    const float* msg_w2    = (const float*)d_in[10];
    const float* msg_b2    = (const float*)d_in[11];
    const float* msg_g2    = (const float*)d_in[12];
    const float* msg_c2    = (const float*)d_in[13];
    const float* upd_w1    = (const float*)d_in[14];
    const float* upd_b1    = (const float*)d_in[15];
    const float* upd_g1    = (const float*)d_in[16];
    const float* upd_c1    = (const float*)d_in[17];
    const float* upd_w2    = (const float*)d_in[18];
    const float* upd_b2    = (const float*)d_in[19];
    const float* upd_g2    = (const float*)d_in[20];
    const float* upd_c2    = (const float*)d_in[21];
    const float* out_w     = (const float*)d_in[22];
    const float* out_b     = (const float*)d_in[23];

    const int n_nodes  = in_sizes[3];
    const int n_edges  = in_sizes[1] / 2;
    const int n_graphs = out_size;
    const int* src = ei;
    const int* dst = ei + n_edges;

    // ---------------- workspace carve (256B aligned)
    auto al = [](size_t v) { return (v + 255) & ~(size_t)255; };
    char* ws = (char*)d_ws;
    size_t off = 0;
    float*          h_f32 = (float*)(ws + off);          off = al(off + (size_t)n_nodes * HID * 4);
    unsigned short* h_bf  = (unsigned short*)(ws + off); off = al(off + (size_t)n_nodes * HID * 2);
    float*          aggr  = (float*)(ws + off);          off = al(off + (size_t)n_nodes * HID * 4);
    float*          sums  = (float*)(ws + off);          off = al(off + (size_t)n_graphs * HID * 4);
    float*          cnt   = (float*)(ws + off);          off = al(off + (size_t)n_graphs * 4);
    unsigned short* wpack = (unsigned short*)(ws + off); // 5 * 13312 ushorts

    // packed layout per layer (ushort offsets)
    const int L_STRIDE = 13312;
    const int OFF_MW1 = 0;      // 160x32 : 10 frags * 512
    const int OFF_MW2 = 5120;   //  32x64 :  4 frags * 512
    const int OFF_UW1 = 7168;   // 128x32 :  8 frags * 512
    const int OFF_UW2 = 11264;  //  32x64 :  4 frags * 512

    // ---------------- pack weights to WMMA fragment layout (bf16)
    for (int l = 0; l < N_LAYERS; ++l) {
        unsigned short* wl = wpack + (size_t)l * L_STRIDE;
        int t1 = 10 * 512, t2 = 4 * 512, t3 = 8 * 512, t4 = 4 * 512;
        pack_w_kernel<<<(t1 + 255) / 256, 256, 0, stream>>>(msg_w1 + (size_t)l * 144 * 32, wl + OFF_MW1, 144, 32, t1);
        pack_w_kernel<<<(t2 + 255) / 256, 256, 0, stream>>>(msg_w2 + (size_t)l * 32 * 64,  wl + OFF_MW2,  32, 64, t2);
        pack_w_kernel<<<(t3 + 255) / 256, 256, 0, stream>>>(upd_w1 + (size_t)l * 128 * 32, wl + OFF_UW1, 128, 32, t3);
        pack_w_kernel<<<(t4 + 255) / 256, 256, 0, stream>>>(upd_w2 + (size_t)l * 32 * 64,  wl + OFF_UW2,  32, 64, t4);
    }

    // ---------------- input projection
    {
        int total = n_nodes * HID;
        lin_in_kernel<<<(total + 255) / 256, 256, 0, stream>>>(x, lin_in_w, lin_in_b, h_f32, h_bf, n_nodes);
    }

    // ---------------- message passing layers
    const int e_tiles = (n_edges + 15) / 16;
    const int n_tiles = (n_nodes + 15) / 16;
    for (int l = 0; l < N_LAYERS; ++l) {
        unsigned short* wl = wpack + (size_t)l * L_STRIDE;
        hipMemsetAsync(aggr, 0, (size_t)n_nodes * HID * sizeof(float), stream);
        edge_msg_kernel<<<(e_tiles + WPB - 1) / WPB, 256, 0, stream>>>(
            src, dst, edge_attr, h_bf, aggr,
            wl + OFF_MW1, wl + OFF_MW2,
            msg_b1 + l * MLP_H, msg_g1 + l * MLP_H, msg_c1 + l * MLP_H,
            msg_b2 + l * HID,   msg_g2 + l * HID,   msg_c2 + l * HID,
            n_edges, e_tiles);
        node_upd_kernel<<<(n_tiles + WPB - 1) / WPB, 256, 0, stream>>>(
            h_f32, h_bf, aggr,
            wl + OFF_UW1, wl + OFF_UW2,
            upd_b1 + l * MLP_H, upd_g1 + l * MLP_H, upd_c1 + l * MLP_H,
            upd_b2 + l * HID,   upd_g2 + l * HID,   upd_c2 + l * HID,
            n_nodes, n_tiles);
    }

    // ---------------- global mean pool + linear head
    hipMemsetAsync(sums, 0, (size_t)n_graphs * HID * sizeof(float), stream);
    hipMemsetAsync(cnt,  0, (size_t)n_graphs * sizeof(float), stream);
    pool_accum_kernel<<<(n_nodes + 255) / 256, 256, 0, stream>>>(h_f32, batch, sums, cnt, n_nodes);
    pool_out_kernel<<<(n_graphs + 255) / 256, 256, 0, stream>>>(sums, cnt, out_w, out_b, (float*)d_out, n_graphs);
}
